// aGCN_69526930588050
// MI455X (gfx1250) — compile-verified, matrix operand order checked by reading, not run on hardware
//
#include <hip/hip_runtime.h>
#include <cstdint>

typedef __attribute__((ext_vector_type(16))) __bf16 v16bf;
typedef __attribute__((ext_vector_type(8)))  float  v8f;

#define N_OBJ 2048
#define N_REL 8192
#define DIM   512

#define BM 128
#define BN 128
#define BK 32
#define BKP 48   // padded LDS row stride in bf16 elements (96B, 16B-aligned)

#define BUF_ELEMS (BM * BKP)                 // one LDS tile buffer (elements)
#define BUF_BYTES (BUF_ELEMS * 2)            // bytes per tile buffer
#define NSTAGE 3                             // async pipeline depth

union Frag    { uint4 u[2]; v16bf v; };
union OutPack { uint4 u; __bf16 h[8]; };

// ---------------------------------------------------------------------------
// One 128x32 x 32x128 tile-step of WMMA work for this wave (4x2 accumulators).
// ---------------------------------------------------------------------------
__device__ __forceinline__ void compute_tile(
    const __bf16* __restrict__ Acur, const __bf16* __restrict__ Bcur,
    int wm, int wn, int lr, int hf, v8f (&acc)[4][2])
{
  // Per 16x32 fragment, lane (lr,hf) loads bf16 K = hf*8..+7 and 16+hf*8..+7
  // -> two b128 LDS loads.
  Frag b[2];
  #pragma unroll
  for (int nt = 0; nt < 2; ++nt) {
    const uint4* rp = (const uint4*)(Bcur + (wn + nt * 16 + lr) * BKP);
    b[nt].u[0] = rp[hf];
    b[nt].u[1] = rp[2 + hf];
  }
  Frag a[4];
  #pragma unroll
  for (int mt = 0; mt < 4; ++mt) {
    const uint4* rp = (const uint4*)(Acur + (wm + mt * 16 + lr) * BKP);
    a[mt].u[0] = rp[hf];
    a[mt].u[1] = rp[2 + hf];
  }
  #pragma unroll
  for (int mt = 0; mt < 4; ++mt)
    #pragma unroll
    for (int nt = 0; nt < 2; ++nt)
      acc[mt][nt] = __builtin_amdgcn_wmma_f32_16x16x32_bf16(
          false, a[mt].v, false, b[nt].v, (short)0, acc[mt][nt],
          false, false);
}

// ---------------------------------------------------------------------------
// Core: accumulate one K-segment of C[M,N] += A[M,K] * Bt[N,K]^T into 4x2
// WMMA f32 accumulators per wave. Block = 256 threads = 8 waves; block tile
// 128x128; wave tile 64x32. Ring of NSTAGE LDS buffers filled by async
// global->LDS loads (ASYNCcnt), branch-free steady-state loop.
// ---------------------------------------------------------------------------
__device__ __forceinline__ void gemm_tiles_seg(
    const __bf16* __restrict__ A, const __bf16* __restrict__ Bt, int K,
    int m0, int n0, __bf16* Asm, __bf16* Bsm, v8f (&acc)[4][2])
{
  const int tid  = threadIdx.x;
  const int lane = tid & 31;
  const int w    = tid >> 5;
  const int wm   = (w & 1) * 64;   // wave M offset in block tile
  const int wn   = (w >> 1) * 32;  // wave N offset in block tile
  const int lr   = lane & 15;      // row (A: M, B: N) within 16
  const int hf   = lane >> 4;      // half-wave selects K group

  // Per-thread staging assignment: 2 b128 chunks of A + 2 of B per tile.
  const int row = tid >> 2;        // 0..63 (and +64 for second chunk)
  const int ch  = tid & 3;         // 4 x 8-bf16 chunks per row
  const __bf16* pa0 = A  + (size_t)(m0 + row) * K + ch * 8;
  const __bf16* pa1 = pa0 + (size_t)64 * K;
  const __bf16* pb0 = Bt + (size_t)(n0 + row) * K + ch * 8;
  const __bf16* pb1 = pb0 + (size_t)64 * K;

  // LDS byte addresses (aperture is 4GB-aligned: low 32 bits == LDS offset)
  const uint32_t abase = (uint32_t)(uintptr_t)Asm;
  const uint32_t bbase = (uint32_t)(uintptr_t)Bsm;
  const uint32_t l0 = (uint32_t)(row * BKP + ch * 8) * 2u;
  const uint32_t l1 = l0 + (uint32_t)(64 * BKP * 2);

  auto issue = [&](uint32_t sel) {
    const uint32_t ao = abase + sel * (uint32_t)BUF_BYTES;
    const uint32_t bo = bbase + sel * (uint32_t)BUF_BYTES;
    asm volatile("global_load_async_to_lds_b128 %0, %1, off"
                 :: "v"(ao + l0), "v"((unsigned long long)(uintptr_t)pa0)
                 : "memory");
    asm volatile("global_load_async_to_lds_b128 %0, %1, off"
                 :: "v"(ao + l1), "v"((unsigned long long)(uintptr_t)pa1)
                 : "memory");
    asm volatile("global_load_async_to_lds_b128 %0, %1, off"
                 :: "v"(bo + l0), "v"((unsigned long long)(uintptr_t)pb0)
                 : "memory");
    asm volatile("global_load_async_to_lds_b128 %0, %1, off"
                 :: "v"(bo + l1), "v"((unsigned long long)(uintptr_t)pb1)
                 : "memory");
    pa0 += BK; pa1 += BK; pb0 += BK; pb1 += BK;   // advance one k-tile
  };

  const int nk = K / BK;   // always >= 16 here, multiple of NSTAGE not needed

  // Prologue: fill the ring (3 tiles, 12 async ops in flight).
  issue(0); issue(1); issue(2);

  // Steady state: branch-free. Waiting to <=8 means the oldest tile's 4 ops
  // have completed (async loads complete in order per wave).
  uint32_t sel = 0;
  for (int kt = 0; kt < nk - NSTAGE; ++kt) {
    asm volatile("s_wait_asynccnt 0x8" ::: "memory");
    __syncthreads();   // all waves' async data visible in LDS
    compute_tile(Asm + sel * BUF_ELEMS, Bsm + sel * BUF_ELEMS,
                 wm, wn, lr, hf, acc);
    __syncthreads();   // WAR: everyone done reading buffer 'sel'
    issue(sel);        // refill the just-freed buffer with tile kt+NSTAGE
    sel = (sel == NSTAGE - 1) ? 0u : sel + 1u;
  }

  // Tail: drain the ring; wait immediates fold to constants when unrolled.
  #pragma unroll
  for (int r = 0; r < NSTAGE; ++r) {
    if (r == 0)      asm volatile("s_wait_asynccnt 0x8" ::: "memory");
    else if (r == 1) asm volatile("s_wait_asynccnt 0x4" ::: "memory");
    else             asm volatile("s_wait_asynccnt 0x0" ::: "memory");
    __syncthreads();
    compute_tile(Asm + sel * BUF_ELEMS, Bsm + sel * BUF_ELEMS,
                 wm, wn, lr, hf, acc);
    __syncthreads();
    sel = (sel == NSTAGE - 1) ? 0u : sel + 1u;
  }
}

// ---------------------------------------------------------------------------
// fc kernel: fcT[n,m] = relu( (A[M,K] * W[N,K]^T)[m,n] + bias[n] ), bf16 out,
// written TRANSPOSED so it directly serves as the Bt operand downstream.
// N is fixed = DIM = 512.
// ---------------------------------------------------------------------------
__global__ __launch_bounds__(256) void k_gemm_fc(
    const __bf16* __restrict__ A, const __bf16* __restrict__ Bt,
    const float* __restrict__ bias, __bf16* __restrict__ fcT,
    int M, int K)
{
  __shared__ __align__(16) __bf16 Asm[NSTAGE * BUF_ELEMS];
  __shared__ __align__(16) __bf16 Bsm[NSTAGE * BUF_ELEMS];
  v8f acc[4][2] = {};

  const int m0 = blockIdx.y * BM;
  const int n0 = blockIdx.x * BN;
  gemm_tiles_seg(A, Bt, K, m0, n0, Asm, Bsm, acc);

  const int tid = threadIdx.x, lane = tid & 31, w = tid >> 5;
  const int wm = (w & 1) * 64, wn = (w >> 1) * 32;
  const int lr = lane & 15, hf = lane >> 4;

  #pragma unroll
  for (int mt = 0; mt < 4; ++mt) {
    #pragma unroll
    for (int nt = 0; nt < 2; ++nt) {
      const int n  = n0 + wn + nt * 16 + lr;
      const float bv = bias[n];
      OutPack p;
      #pragma unroll
      for (int v = 0; v < 8; ++v)
        p.h[v] = (__bf16)fmaxf(acc[mt][nt][v] + bv, 0.0f);
      const int m = m0 + wm + mt * 16 + hf * 8;   // 8 consecutive m's
      *(uint4*)(fcT + (size_t)n * M + m) = p.u;
    }
  }
}

// ---------------------------------------------------------------------------
// update kernel: out = residual + scale * sum_seg( A_s[M,Ks] * Bt_s[512,Ks]^T )
// Emits fp32 (next-step residual / final projection) + bf16 copy
// (next-step fc input). N fixed = 512.
// ---------------------------------------------------------------------------
struct Segs {
  const __bf16* A0; const __bf16* B0; int K0;
  const __bf16* A1; const __bf16* B1; int K1;
  const __bf16* A2; const __bf16* B2; int K2;
  int nseg;
};

__global__ __launch_bounds__(256) void k_gemm_update(
    Segs s, const float* __restrict__ res, float scale,
    float* __restrict__ outf, __bf16* __restrict__ outb, int M)
{
  __shared__ __align__(16) __bf16 Asm[NSTAGE * BUF_ELEMS];
  __shared__ __align__(16) __bf16 Bsm[NSTAGE * BUF_ELEMS];
  v8f acc[4][2] = {};

  const int m0 = blockIdx.y * BM;
  const int n0 = blockIdx.x * BN;

  gemm_tiles_seg(s.A0, s.B0, s.K0, m0, n0, Asm, Bsm, acc);
  if (s.nseg > 1) gemm_tiles_seg(s.A1, s.B1, s.K1, m0, n0, Asm, Bsm, acc);
  if (s.nseg > 2) gemm_tiles_seg(s.A2, s.B2, s.K2, m0, n0, Asm, Bsm, acc);

  const int tid = threadIdx.x, lane = tid & 31, w = tid >> 5;
  const int wm = (w & 1) * 64, wn = (w >> 1) * 32;
  const int lr = lane & 15, hf = lane >> 4;

  #pragma unroll
  for (int mt = 0; mt < 4; ++mt) {
    #pragma unroll
    for (int nt = 0; nt < 2; ++nt) {
      const int n = n0 + wn + nt * 16 + lr;
      #pragma unroll
      for (int v = 0; v < 8; ++v) {
        const int m = m0 + wm + mt * 16 + hf * 8 + v;
        const size_t idx = (size_t)m * DIM + n;
        const float val = res[idx] + acc[mt][nt][v] * scale;
        outf[idx] = val;
        outb[idx] = (__bf16)val;
      }
    }
  }
}

// ---------------------------------------------------------------------------
// Precompute / utility kernels
// ---------------------------------------------------------------------------
__global__ void k_f32_to_bf16(const float* __restrict__ src,
                              __bf16* __restrict__ dst, int n)
{
  int i = blockIdx.x * blockDim.x + threadIdx.x;
  if (i < n) dst[i] = (__bf16)src[i];
}

// Row-normalized bf16 copy: dst[r,c] = src[r,c] / (rowsum(r) + eps)
__global__ __launch_bounds__(256) void k_row_norm(
    const float* __restrict__ src, __bf16* __restrict__ dst, int C)
{
  const int row = blockIdx.x;
  const float* s = src + (size_t)row * C;
  __bf16* d = dst + (size_t)row * C;
  __shared__ float red[256];
  float acc = 0.0f;
  for (int c = threadIdx.x; c < C; c += 256) acc += s[c];
  red[threadIdx.x] = acc;
  __syncthreads();
  for (int off = 128; off > 0; off >>= 1) {
    if (threadIdx.x < off) red[threadIdx.x] += red[threadIdx.x + off];
    __syncthreads();
  }
  const float sc = 1.0f / (red[0] + 1e-7f);
  for (int c = threadIdx.x; c < C; c += 256) d[c] = (__bf16)(s[c] * sc);
}

// Per-column reciprocal sums: scale[c] = 1/(sum_r src[r,c] + eps)
__global__ void k_col_scale(const float* __restrict__ src,
                            float* __restrict__ scale, int R, int C)
{
  int c = blockIdx.x * blockDim.x + threadIdx.x;
  if (c >= C) return;
  float s = 0.0f;
  for (int r = 0; r < R; ++r) s += src[(size_t)r * C + c];
  scale[c] = 1.0f / (s + 1e-7f);
}

// Column-normalized transpose: dst[c,r] = src[r,c] * csc[c], bf16 out.
__global__ __launch_bounds__(256) void k_transpose_scale(
    const float* __restrict__ src, const float* __restrict__ csc,
    __bf16* __restrict__ dst, int R, int C)
{
  __shared__ float tile[32][33];
  const int c0 = blockIdx.x * 32, r0 = blockIdx.y * 32;
  const int tx = threadIdx.x & 31, ty = threadIdx.x >> 5;  // ty: 0..7
  #pragma unroll
  for (int j = 0; j < 32; j += 8)
    tile[ty + j][tx] = src[(size_t)(r0 + ty + j) * C + c0 + tx];
  __syncthreads();
  #pragma unroll
  for (int j = 0; j < 32; j += 8) {
    const int c = c0 + ty + j;
    dst[(size_t)c * R + r0 + tx] = (__bf16)(tile[tx][ty + j] * csc[c]);
  }
}

// Tiny final projections in fp32: out[m,j] = feat[m,:] . W[j,:] + b[j]
__global__ void k_proj(const float* __restrict__ feat,
                       const float* __restrict__ W,
                       const float* __restrict__ b,
                       float* __restrict__ out, int M, int J)
{
  int idx = blockIdx.x * blockDim.x + threadIdx.x;
  if (idx >= M * J) return;
  const int m = idx / J, j = idx % J;
  const float* f = feat + (size_t)m * DIM;
  const float* w = W + (size_t)j * DIM;
  float s = 0.0f;
  for (int k = 0; k < DIM; ++k) s += f[k] * w[k];
  out[idx] = s + b[j];
}

// ---------------------------------------------------------------------------
// Host orchestration
// ---------------------------------------------------------------------------
extern "C" void kernel_launch(void* const* d_in, const int* in_sizes, int n_in,
                              void* d_out, int out_size, void* d_ws, size_t ws_size,
                              hipStream_t stream)
{
  const float* obj_in = (const float*)d_in[0];   // [2048,512]
  const float* rel_in = (const float*)d_in[1];   // [8192,512]
  const float* obj_obj = (const float*)d_in[2];  // [2048,2048]
  const float* sub_rel = (const float*)d_in[3];  // [2048,8192]
  const float* obj_rel = (const float*)d_in[4];  // [2048,8192]
  const float* Wf[5] = { (const float*)d_in[5], (const float*)d_in[7],
                         (const float*)d_in[9], (const float*)d_in[11],
                         (const float*)d_in[13] };
  const float* bf_[5] = { (const float*)d_in[6], (const float*)d_in[8],
                          (const float*)d_in[10], (const float*)d_in[12],
                          (const float*)d_in[14] };
  const float* Wobj = (const float*)d_in[15];
  const float* bobj = (const float*)d_in[16];
  const float* Wrel = (const float*)d_in[17];
  const float* brel = (const float*)d_in[18];

  // Bump allocator over d_ws (256B aligned carves).
  char* ws = (char*)d_ws;
  size_t off = 0;
  auto alloc = [&](size_t bytes) -> void* {
    void* p = ws + off;
    off += (bytes + 255) & ~(size_t)255;
    return p;
  };

  __bf16* oo_n  = (__bf16*)alloc((size_t)N_OBJ * N_OBJ * 2);
  __bf16* sr_n  = (__bf16*)alloc((size_t)N_OBJ * N_REL * 2);
  __bf16* or_n  = (__bf16*)alloc((size_t)N_OBJ * N_REL * 2);
  __bf16* sr_tn = (__bf16*)alloc((size_t)N_REL * N_OBJ * 2);
  __bf16* or_tn = (__bf16*)alloc((size_t)N_REL * N_OBJ * 2);
  float*  csc_sr = (float*)alloc((size_t)N_REL * 4);
  float*  csc_or = (float*)alloc((size_t)N_REL * 4);
  __bf16* Wb[5];
  for (int i = 0; i < 5; ++i) Wb[i] = (__bf16*)alloc((size_t)DIM * DIM * 2);
  __bf16* OB[2] = { (__bf16*)alloc((size_t)N_OBJ * DIM * 2),
                    (__bf16*)alloc((size_t)N_OBJ * DIM * 2) };
  __bf16* RB[2] = { (__bf16*)alloc((size_t)N_REL * DIM * 2),
                    (__bf16*)alloc((size_t)N_REL * DIM * 2) };
  float* Of[2] = { (float*)alloc((size_t)N_OBJ * DIM * 4),
                   (float*)alloc((size_t)N_OBJ * DIM * 4) };
  float* Rf[2] = { (float*)alloc((size_t)N_REL * DIM * 4),
                   (float*)alloc((size_t)N_REL * DIM * 4) };
  __bf16* fc0T = (__bf16*)alloc((size_t)DIM * N_REL * 2);
  __bf16* fc1T = (__bf16*)alloc((size_t)DIM * N_REL * 2);
  __bf16* fc2T = (__bf16*)alloc((size_t)DIM * N_OBJ * 2);
  __bf16* fc3T = (__bf16*)alloc((size_t)DIM * N_OBJ * 2);
  __bf16* fc4T = (__bf16*)alloc((size_t)DIM * N_OBJ * 2);

  // --- One-time precompute (attention matrices are step-invariant) ---
  for (int i = 0; i < 5; ++i) {
    int n = DIM * DIM;
    k_f32_to_bf16<<<(n + 255) / 256, 256, 0, stream>>>(Wf[i], Wb[i], n);
  }
  k_row_norm<<<N_OBJ, 256, 0, stream>>>(obj_obj, oo_n, N_OBJ);
  k_row_norm<<<N_OBJ, 256, 0, stream>>>(sub_rel, sr_n, N_REL);
  k_row_norm<<<N_OBJ, 256, 0, stream>>>(obj_rel, or_n, N_REL);
  k_col_scale<<<N_REL / 256, 256, 0, stream>>>(sub_rel, csc_sr, N_OBJ, N_REL);
  k_col_scale<<<N_REL / 256, 256, 0, stream>>>(obj_rel, csc_or, N_OBJ, N_REL);
  {
    dim3 g(N_REL / 32, N_OBJ / 32);
    k_transpose_scale<<<g, 256, 0, stream>>>(sub_rel, csc_sr, sr_tn, N_OBJ, N_REL);
    k_transpose_scale<<<g, 256, 0, stream>>>(obj_rel, csc_or, or_tn, N_OBJ, N_REL);
  }
  {
    int n = N_OBJ * DIM;
    k_f32_to_bf16<<<(n + 255) / 256, 256, 0, stream>>>(obj_in, OB[0], n);
    n = N_REL * DIM;
    k_f32_to_bf16<<<(n + 255) / 256, 256, 0, stream>>>(rel_in, RB[0], n);
  }

  auto fc = [&](const __bf16* A, const __bf16* Bt, const float* bias,
                __bf16* out, int M, int K) {
    dim3 g(DIM / BN, M / BM);
    k_gemm_fc<<<g, 256, 0, stream>>>(A, Bt, bias, out, M, K);
  };

  // --- 3 message-passing steps ---
  int cur = 0;
  for (int t = 0; t < 3; ++t) {
    const int nxt = cur ^ 1;
    const float* resO = (t == 0) ? obj_in : Of[cur];
    const float* resR = (t == 0) ? rel_in : Rf[cur];

    fc(OB[cur], Wb[4], bf_[4], fc4T, N_OBJ, DIM);   // obj -> fc4
    fc(RB[cur], Wb[0], bf_[0], fc0T, N_REL, DIM);   // rel -> fc0
    fc(RB[cur], Wb[1], bf_[1], fc1T, N_REL, DIM);   // rel -> fc1
    fc(OB[cur], Wb[2], bf_[2], fc2T, N_OBJ, DIM);   // obj -> fc2
    fc(OB[cur], Wb[3], bf_[3], fc3T, N_OBJ, DIM);   // obj -> fc3

    Segs so = { oo_n, fc4T, N_OBJ,
                sr_n, fc0T, N_REL,
                or_n, fc1T, N_REL, 3 };
    k_gemm_update<<<dim3(DIM / BN, N_OBJ / BM), 256, 0, stream>>>(
        so, resO, 1.0f / 3.0f, Of[nxt], OB[nxt], N_OBJ);

    Segs sre = { sr_tn, fc2T, N_OBJ,
                 or_tn, fc3T, N_OBJ,
                 nullptr, nullptr, 0, 2 };
    k_gemm_update<<<dim3(DIM / BN, N_REL / BM), 256, 0, stream>>>(
        sre, resR, 0.5f, Rf[nxt], RB[nxt], N_REL);

    cur = nxt;
  }

  // --- Final projections into d_out (obj_dist flat, then rel_dist flat) ---
  float* out = (float*)d_out;
  {
    int n = N_OBJ * 26;
    k_proj<<<(n + 255) / 256, 256, 0, stream>>>(Of[cur], Wobj, bobj, out, N_OBJ, 26);
    int n2 = N_REL * 17;
    k_proj<<<(n2 + 255) / 256, 256, 0, stream>>>(Rf[cur], Wrel, brel,
                                                 out + (size_t)N_OBJ * 26, N_REL, 17);
  }
}
